// HGTModule_74156905333027
// MI455X (gfx1250) — matrix operand order repaired and checked by reading, not compiled, not run on hardware
//
#include <hip/hip_runtime.h>
#include <hip/hip_bf16.h>

typedef __attribute__((ext_vector_type(16))) __bf16 v16bf;
typedef __attribute__((ext_vector_type(8)))  float  v8f;
typedef __attribute__((ext_vector_type(4)))  unsigned int u32x4;
typedef unsigned int u32;

union FragA { v16bf v; u32x4 q[2]; };
union BF2   { u32 u; __bf16 h[2]; };
union H8    { u32x4 q; __bf16 h[8]; };

// ---------------------------------------------------------------------------
// Weight pack: fuse optional per-head relation matrix R[4,64,64] into W[256,256]
// and emit bf16 pairs in the contiguous-k B-fragment layout gemm_bf16 uses:
//   Wp[((kt*16+nt)*32+lane)*8+j] packs K = kt*32 + 16*(lane>>4) + 2j (and +1),
//   N = nt*16 + (lane&15).
// ---------------------------------------------------------------------------
__global__ void pack_weight(const float* __restrict__ W, const float* __restrict__ R,
                            u32* __restrict__ Wp) {
  int t = blockIdx.x * blockDim.x + threadIdx.x;
  if (t >= 8 * 16 * 32 * 8) return;
  int j    = t & 7;
  int lane = (t >> 3) & 31;
  int nt   = (t >> 8) & 15;
  int kt   = (t >> 12) & 7;
  int n  = nt * 16 + (lane & 15);
  int k0 = kt * 32 + ((lane >> 4) * 16) + 2 * j;
  float w0, w1;
  if (R) {
    int h = n >> 6, nd = n & 63;
    const float* Rh = R + (size_t)h * 64 * 64 + nd;  // R[h,d,nd], stride 64 over d
    const float* W0 = W + (size_t)k0 * 256 + h * 64;
    const float* W1 = W0 + 256;
    float s0 = 0.f, s1 = 0.f;
    for (int d = 0; d < 64; ++d) {
      float r = Rh[(size_t)d * 64];
      s0 += W0[d] * r;
      s1 += W1[d] * r;
    }
    w0 = s0; w1 = s1;
  } else {
    w0 = W[(size_t)k0 * 256 + n];
    w1 = W[(size_t)(k0 + 1) * 256 + n];
  }
  BF2 p; p.h[0] = (__bf16)w0; p.h[1] = (__bf16)w1;
  Wp[((size_t)(kt * 16 + nt) * 32 + lane) * 8 + j] = p.u;
}

__global__ void fuse_bias(const float* __restrict__ b, const float* __restrict__ R,
                          float* __restrict__ out) {
  int n = blockIdx.x * blockDim.x + threadIdx.x;
  if (n >= 256) return;
  if (R) {
    int h = n >> 6, nd = n & 63;
    float s = 0.f;
    for (int d = 0; d < 64; ++d) s += b[h * 64 + d] * R[(size_t)h * 4096 + (size_t)d * 64 + nd];
    out[n] = s;
  } else {
    out[n] = b[n];
  }
}

__global__ void cvt_bf16(const float* __restrict__ x, __bf16* __restrict__ y, int n) {
  int i = blockIdx.x * blockDim.x + threadIdx.x;
  if (i < n) y[i] = (__bf16)x[i];
}

__global__ void fill_f32(float* __restrict__ p, float v, int n) {
  int i = blockIdx.x * blockDim.x + threadIdx.x;
  if (i < n) p[i] = v;
}

// ---------------------------------------------------------------------------
// GEMM: C[N,256] = A_bf16[N,256] @ W(packed) + bias, optional skip epilogue.
// Block = 8 waves; wave computes 32 rows x 64 cols (2 A-frags, 4 B-frags,
// 8 v_wmma_f32_16x16x32_bf16 per k-step, 8 k-steps over K=256).
// All accumulators are named scalars of v8f so nothing can spill via dynamic
// indexing; the k-loop is branch-free (tile-1 rows alias tile-0 when OOB and
// only the stores are guarded), so WMMA always runs with full EXEC.
// ---------------------------------------------------------------------------
__global__ __launch_bounds__(256)
void gemm_bf16(const __bf16* __restrict__ A, const u32* __restrict__ Wp,
               const float* __restrict__ bias, int N,
               float* __restrict__ outF, __bf16* __restrict__ outH,
               int epilogue, const float* __restrict__ skipGate,
               const float* __restrict__ xOld) {
  const int wave = threadIdx.x >> 5;
  const int lane = threadIdx.x & 31;
  const int hl   = lane >> 4;    // K half selector
  const int mr   = lane & 15;    // row-in-tile / col-in-tile
  const int m0   = blockIdx.x * 256 + wave * 32;
  if (m0 >= N) return;
  const bool hasT1 = (m0 + 16) < N;
  const int ntBase = blockIdx.y * 4;

  const __bf16* arow0 = A + (size_t)(m0 + mr) * 256;
  const __bf16* arow1 = hasT1 ? (arow0 + 16 * 256) : arow0;  // safe alias when OOB

  v8f c0 = {}, c1 = {}, c2 = {}, c3 = {}, c4 = {}, c5 = {}, c6 = {}, c7 = {};

#pragma unroll
  for (int kt = 0; kt < 8; ++kt) {
    const int kb = kt * 32 + hl * 16;
    FragA a0, a1;
    a0.q[0] = *(const u32x4*)(arow0 + kb);
    a0.q[1] = *(const u32x4*)(arow0 + kb + 8);
    a1.q[0] = *(const u32x4*)(arow1 + kb);
    a1.q[1] = *(const u32x4*)(arow1 + kb + 8);
    if (kt < 7) __builtin_prefetch(arow0 + kb + 32, 0, 3);

    const u32* wp = Wp + ((size_t)(kt * 16 + ntBase) * 32 + lane) * 8;
    FragA b0, b1, b2, b3;
    b0.q[0] = *(const u32x4*)(wp + 0);   b0.q[1] = *(const u32x4*)(wp + 4);
    b1.q[0] = *(const u32x4*)(wp + 256); b1.q[1] = *(const u32x4*)(wp + 260);
    b2.q[0] = *(const u32x4*)(wp + 512); b2.q[1] = *(const u32x4*)(wp + 516);
    b3.q[0] = *(const u32x4*)(wp + 768); b3.q[1] = *(const u32x4*)(wp + 772);

    c0 = __builtin_amdgcn_wmma_f32_16x16x32_bf16(false, a0.v, false, b0.v, (short)0, c0, false, false);
    c4 = __builtin_amdgcn_wmma_f32_16x16x32_bf16(false, a1.v, false, b0.v, (short)0, c4, false, false);
    c1 = __builtin_amdgcn_wmma_f32_16x16x32_bf16(false, a0.v, false, b1.v, (short)0, c1, false, false);
    c5 = __builtin_amdgcn_wmma_f32_16x16x32_bf16(false, a1.v, false, b1.v, (short)0, c5, false, false);
    c2 = __builtin_amdgcn_wmma_f32_16x16x32_bf16(false, a0.v, false, b2.v, (short)0, c2, false, false);
    c6 = __builtin_amdgcn_wmma_f32_16x16x32_bf16(false, a1.v, false, b2.v, (short)0, c6, false, false);
    c3 = __builtin_amdgcn_wmma_f32_16x16x32_bf16(false, a0.v, false, b3.v, (short)0, c3, false, false);
    c7 = __builtin_amdgcn_wmma_f32_16x16x32_bf16(false, a1.v, false, b3.v, (short)0, c7, false, false);
  }

  float gate = 1.f, og = 0.f;
  if (epilogue == 1) {
    float s = skipGate[0];
    gate = 1.f / (1.f + __expf(-s));
    og = 1.f - gate;
  }

  auto storeTile = [&](v8f cc, int i, int rowBase) {
    const int ncol = (ntBase + i) * 16 + mr;
    const float bb = bias[ncol];
#pragma unroll
    for (int g = 0; g < 8; ++g) {
      int row = rowBase + g + 8 * hl;   // C/D frag: VGPR g, lanes16-31 -> M+8
      float v = cc[g] + bb;
      size_t idx = (size_t)row * 256 + ncol;
      if (epilogue == 1) {
        float r = gate * v + og * xOld[idx];
        outF[idx] = r;
        outH[idx] = (__bf16)r;
      } else {
        if (outF) outF[idx] = v;
        if (outH) outH[idx] = (__bf16)v;
      }
    }
  };
  storeTile(c0, 0, m0);
  storeTile(c1, 1, m0);
  storeTile(c2, 2, m0);
  storeTile(c3, 3, m0);
  if (hasT1) {
    storeTile(c4, 0, m0 + 16);
    storeTile(c5, 1, m0 + 16);
    storeTile(c6, 2, m0 + 16);
    storeTile(c7, 3, m0 + 16);
  }
}

// ---------------------------------------------------------------------------
// Edge-phase kernels (wave32: 1 wave per edge; head = 8 lanes x 8 dims)
// ---------------------------------------------------------------------------
__device__ __forceinline__ void atomicMaxFloat(float* addr, float v) {
  if (v >= 0.f) atomicMax((int*)addr, __float_as_int(v));
  else          atomicMin((unsigned int*)addr, __float_as_uint(v));
}

__global__ __launch_bounds__(256)
void edge_alpha(const int* __restrict__ src, const int* __restrict__ dst, int E,
                const __bf16* __restrict__ Q, const __bf16* __restrict__ K,
                const float* __restrict__ p_rel,
                float* __restrict__ alphaBuf, float* __restrict__ maxBuf) {
  int wave = threadIdx.x >> 5, lane = threadIdx.x & 31;
  int e = blockIdx.x * 8 + wave;
  if (e >= E) return;
  int s = src[e], d = dst[e];
  int h = lane >> 3, t = lane & 7;
  H8 qv, kv;
  qv.q = *(const u32x4*)(Q + (size_t)d * 256 + h * 64 + t * 8);
  kv.q = *(const u32x4*)(K + (size_t)s * 256 + h * 64 + t * 8);
  float acc = 0.f;
#pragma unroll
  for (int i = 0; i < 8; ++i) acc += (float)qv.h[i] * (float)kv.h[i];
  acc += __shfl_xor(acc, 1, 32);
  acc += __shfl_xor(acc, 2, 32);
  acc += __shfl_xor(acc, 4, 32);
  if (t == 0) {
    float alpha = acc * p_rel[h] * 0.125f;  // scale = 1/sqrt(64)
    alphaBuf[(size_t)e * 4 + h] = alpha;
    atomicMaxFloat(maxBuf + (size_t)d * 4 + h, alpha);
  }
}

__global__ void edge_exp(const int* __restrict__ dst, int E,
                         float* __restrict__ alphaBuf, const float* __restrict__ maxBuf,
                         float* __restrict__ denBuf) {
  int i = blockIdx.x * blockDim.x + threadIdx.x;
  if (i >= E * 4) return;
  int e = i >> 2, h = i & 3;
  int d = dst[e];
  float ex = __expf(alphaBuf[i] - maxBuf[(size_t)d * 4 + h]);
  alphaBuf[i] = ex;
  atomicAdd(denBuf + (size_t)d * 4 + h, ex);
}

__global__ __launch_bounds__(256)
void edge_scatter(const int* __restrict__ src, const int* __restrict__ dst, int E,
                  const __bf16* __restrict__ V, const float* __restrict__ alphaBuf,
                  const float* __restrict__ denBuf, float* __restrict__ agg) {
  int wave = threadIdx.x >> 5, lane = threadIdx.x & 31;
  int e = blockIdx.x * 8 + wave;
  if (e >= E) return;
  int s = src[e], d = dst[e];
  int h = lane >> 3, t = lane & 7;
  float w = alphaBuf[(size_t)e * 4 + h] / (denBuf[(size_t)d * 4 + h] + 1e-16f);
  H8 vv;
  vv.q = *(const u32x4*)(V + (size_t)s * 256 + h * 64 + t * 8);
  float* ap = agg + (size_t)d * 256 + h * 64 + t * 8;
#pragma unroll
  for (int i = 0; i < 8; ++i) atomicAdd(ap + i, w * (float)vv.h[i]);
}

__global__ void mean_gelu(const float* __restrict__ agg, float invCnt,
                          __bf16* __restrict__ out, int n) {
  int i = blockIdx.x * blockDim.x + threadIdx.x;
  if (i >= n) return;
  float x = agg[i] * invCnt;
  float g = 0.5f * x * (1.f + erff(x * 0.70710678118654752f));
  out[i] = (__bf16)g;
}

__global__ void write_out(const float* __restrict__ xA, const float* __restrict__ xP,
                          const int* __restrict__ bA, float* __restrict__ out, int total) {
  int i = blockIdx.x * blockDim.x + threadIdx.x;
  if (i >= total) return;
  int nA = bA[0] * 256;
  out[i] = (i < nA) ? xA[i] : xP[i - nA];
}

// ---------------------------------------------------------------------------
extern "C" void kernel_launch(void* const* d_in, const int* in_sizes, int n_in,
                              void* d_out, int out_size, void* d_ws, size_t ws_size,
                              hipStream_t stream) {
  (void)n_in; (void)ws_size;
  const float* xA_in = (const float*)d_in[0];
  const float* xP_in = (const float*)d_in[1];
  const int* eSrc[3] = { (const int*)d_in[2], (const int*)d_in[4], (const int*)d_in[6] };
  const int* eDst[3] = { (const int*)d_in[3], (const int*)d_in[5], (const int*)d_in[7] };
  const int* bA = (const int*)d_in[8];
  const int  Ecnt[3] = { in_sizes[2], in_sizes[4], in_sizes[6] };
  const int  NA = in_sizes[0] / 256;
  const int  NP = in_sizes[1] / 256;

  // ---- param pointer table; detect pytree flatten order (sorted vs insertion)
  auto F = [&](int i) -> const float* { return (const float*)d_in[i]; };
  bool sorted = (in_sizes[10] == 16384);  // a_rel first (jax tree) vs kW first (insertion)
  struct NodeP { const float *kW,*kb,*qW,*qb,*vW,*vb,*aW,*ab,*skip; };
  struct EdgeP { const float *a_rel,*m_rel,*p_rel; };
  NodeP npar[2][2];   // [layer][0=author,1=paper]
  EdgeP edp[2][3];    // [layer][0=writes,1=rev_writes,2=cites]
  for (int L = 0; L < 2; ++L) {
    int b0 = 10 + L * 27;
    if (sorted) {
      const int em[3] = {2, 1, 0};  // sorted blocks: cites, rev_writes, writes
      for (int r = 0; r < 3; ++r) {
        int bb = b0 + em[r] * 3;
        edp[L][r] = { F(bb), F(bb + 1), F(bb + 2) };
      }
      for (int t = 0; t < 2; ++t) {
        int nb = b0 + 9 + t * 9;  // aW,ab,kW,kb,qW,qb,skip,vW,vb
        npar[L][t].aW = F(nb);     npar[L][t].ab = F(nb + 1);
        npar[L][t].kW = F(nb + 2); npar[L][t].kb = F(nb + 3);
        npar[L][t].qW = F(nb + 4); npar[L][t].qb = F(nb + 5);
        npar[L][t].skip = F(nb + 6);
        npar[L][t].vW = F(nb + 7); npar[L][t].vb = F(nb + 8);
      }
    } else {
      for (int t = 0; t < 2; ++t) {
        int nb = b0 + t * 9;      // kW,kb,qW,qb,vW,vb,aW,ab,skip
        npar[L][t].kW = F(nb);     npar[L][t].kb = F(nb + 1);
        npar[L][t].qW = F(nb + 2); npar[L][t].qb = F(nb + 3);
        npar[L][t].vW = F(nb + 4); npar[L][t].vb = F(nb + 5);
        npar[L][t].aW = F(nb + 6); npar[L][t].ab = F(nb + 7);
        npar[L][t].skip = F(nb + 8);
      }
      for (int r = 0; r < 3; ++r) {
        int bb = b0 + 18 + r * 3;  // writes, rev_writes, cites
        edp[L][r] = { F(bb), F(bb + 1), F(bb + 2) };
      }
    }
  }

  // ---- workspace carve-out
  char* base = (char*)d_ws;
  size_t off = 0;
  auto alloc = [&](size_t bytes) -> void* {
    void* p = base + off;
    off += (bytes + 255) & ~(size_t)255;
    return p;
  };
  int maxE = Ecnt[0] > Ecnt[1] ? Ecnt[0] : Ecnt[1];
  if (Ecnt[2] > maxE) maxE = Ecnt[2];
  float*  xAf  = (float*) alloc((size_t)NA * 256 * 4);
  float*  xPf  = (float*) alloc((size_t)NP * 256 * 4);
  __bf16* xAh  = (__bf16*)alloc((size_t)NA * 256 * 2);
  __bf16* xPh  = (__bf16*)alloc((size_t)NP * 256 * 2);
  __bf16* QAh  = (__bf16*)alloc((size_t)NA * 256 * 2);  // reused as gelu buf
  __bf16* QPh  = (__bf16*)alloc((size_t)NP * 256 * 2);  // reused as gelu buf
  __bf16* Kh   = (__bf16*)alloc((size_t)NP * 256 * 2);
  __bf16* Vh   = (__bf16*)alloc((size_t)NP * 256 * 2);
  float*  aggA = (float*) alloc((size_t)NA * 256 * 4);
  float*  aggP = (float*) alloc((size_t)NP * 256 * 4);
  float*  alphaB = (float*)alloc((size_t)maxE * 4 * 4);
  float*  maxB   = (float*)alloc((size_t)NP * 4 * 4);
  float*  denB   = (float*)alloc((size_t)NP * 4 * 4);
  u32*    wpk    = (u32*)  alloc((size_t)20 * 32768 * 4);
  float*  bpk    = (float*)alloc((size_t)20 * 256 * 4);
  auto WP = [&](int L, int i) { return wpk + ((size_t)L * 10 + i) * 32768; };
  auto BP = [&](int L, int i) { return bpk + ((size_t)L * 10 + i) * 256; };

  auto gemm = [&](const __bf16* A, const u32* Wpp, const float* bias, int N,
                  float* oF, __bf16* oH, int ep, const float* sk, const float* xo) {
    dim3 g((N + 255) / 256, 4);
    gemm_bf16<<<g, 256, 0, stream>>>(A, Wpp, bias, N, oF, oH, ep, sk, xo);
  };

  // ---- 1. pack weights + biases for both layers
  // index: 0 qA, 1 qP, 2 Kw, 3 Vw, 4 Kr, 5 Vr, 6 Kc, 7 Vc, 8 aA, 9 aP
  for (int L = 0; L < 2; ++L) {
    const float* Ws[10] = { npar[L][0].qW, npar[L][1].qW,
                            npar[L][0].kW, npar[L][0].vW,
                            npar[L][1].kW, npar[L][1].vW,
                            npar[L][1].kW, npar[L][1].vW,
                            npar[L][0].aW, npar[L][1].aW };
    const float* Rs[10] = { nullptr, nullptr,
                            edp[L][0].a_rel, edp[L][0].m_rel,
                            edp[L][1].a_rel, edp[L][1].m_rel,
                            edp[L][2].a_rel, edp[L][2].m_rel,
                            nullptr, nullptr };
    const float* Bs[10] = { npar[L][0].qb, npar[L][1].qb,
                            npar[L][0].kb, npar[L][0].vb,
                            npar[L][1].kb, npar[L][1].vb,
                            npar[L][1].kb, npar[L][1].vb,
                            npar[L][0].ab, npar[L][1].ab };
    for (int i = 0; i < 10; ++i) {
      pack_weight<<<128, 256, 0, stream>>>(Ws[i], Rs[i], WP(L, i));
      fuse_bias<<<1, 256, 0, stream>>>(Bs[i], Rs[i], BP(L, i));
    }
  }

  // ---- 2. initial fp32 -> bf16 activation conversion
  cvt_bf16<<<(NA * 256 + 255) / 256, 256, 0, stream>>>(xA_in, xAh, NA * 256);
  cvt_bf16<<<(NP * 256 + 255) / 256, 256, 0, stream>>>(xP_in, xPh, NP * 256);

  const float* curAf = xA_in;
  const float* curPf = xP_in;

  for (int L = 0; L < 2; ++L) {
    // Q projections (bf16 out)
    gemm(xAh, WP(L, 0), BP(L, 0), NA, nullptr, QAh, 0, nullptr, nullptr);
    gemm(xPh, WP(L, 1), BP(L, 1), NP, nullptr, QPh, 0, nullptr, nullptr);

    fill_f32<<<(NA * 256 + 255) / 256, 256, 0, stream>>>(aggA, 0.f, NA * 256);
    fill_f32<<<(NP * 256 + 255) / 256, 256, 0, stream>>>(aggP, 0.f, NP * 256);

    // per edge type: relation K/V GEMMs then 3-pass segment-softmax attention
    const __bf16* srcXs[3] = { xAh, xPh, xPh };
    const int     srcNs[3] = { NA, NP, NP };
    const __bf16* Qd[3]    = { QPh, QAh, QPh };
    const int     nDs[3]   = { NP, NA, NP };
    float*        aggD[3]  = { aggP, aggA, aggP };
    const int     kIdx[3]  = { 2, 4, 6 };
    for (int r = 0; r < 3; ++r) {
      int E = Ecnt[r], nD = nDs[r];
      gemm(srcXs[r], WP(L, kIdx[r]),     BP(L, kIdx[r]),     srcNs[r], nullptr, Kh, 0, nullptr, nullptr);
      gemm(srcXs[r], WP(L, kIdx[r] + 1), BP(L, kIdx[r] + 1), srcNs[r], nullptr, Vh, 0, nullptr, nullptr);
      fill_f32<<<(nD * 4 + 255) / 256, 256, 0, stream>>>(maxB, -__builtin_inff(), nD * 4);
      fill_f32<<<(nD * 4 + 255) / 256, 256, 0, stream>>>(denB, 0.f, nD * 4);
      edge_alpha<<<(E + 7) / 8, 256, 0, stream>>>(eSrc[r], eDst[r], E, Qd[r], Kh,
                                                  edp[L][r].p_rel, alphaB, maxB);
      edge_exp<<<(E * 4 + 255) / 256, 256, 0, stream>>>(eDst[r], E, alphaB, maxB, denB);
      edge_scatter<<<(E + 7) / 8, 256, 0, stream>>>(eSrc[r], eDst[r], E, Vh, alphaB, denB, aggD[r]);
    }

    // mean over edge types (author: 1, paper: 2) + exact GELU -> bf16 (reuse Q bufs)
    mean_gelu<<<(NA * 256 + 255) / 256, 256, 0, stream>>>(aggA, 1.0f, QAh, NA * 256);
    mean_gelu<<<(NP * 256 + 255) / 256, 256, 0, stream>>>(aggP, 0.5f, QPh, NP * 256);

    // output projection + learnable skip gate; writes fp32 (residual) + bf16 (next layer)
    gemm(QAh, WP(L, 8), BP(L, 8), NA, xAf, xAh, 1, npar[L][0].skip, curAf);
    gemm(QPh, WP(L, 9), BP(L, 9), NP, xPf, xPh, 1, npar[L][1].skip, curPf);
    curAf = xAf;
    curPf = xPf;
  }

  write_out<<<(out_size + 255) / 256, 256, 0, stream>>>(xAf, xPf, bA, (float*)d_out, out_size);
}
